// Transformer_73237782332082
// MI455X (gfx1250) — matrix operand order repaired and checked by reading, not compiled, not run on hardware
//
#include <hip/hip_runtime.h>
#include <cstdint>

// ---------------------------------------------------------------------------
// Types for CDNA5 WMMA (wave32): V_WMMA_F32_16X16X32_BF16
// ---------------------------------------------------------------------------
typedef __attribute__((ext_vector_type(16))) __bf16 v16bf;
typedef __attribute__((ext_vector_type(8)))  float  v8f;
typedef __attribute__((ext_vector_type(4)))  int    v4i;

union Frag {              // 16x32 bf16 A/B fragment: 8 VGPRs
    v16bf v;
    uint4 q[2];
};
union BF8 {               // 8 packed bf16 loaded as one b128
    uint4 q;
    unsigned short e[8];
};

__device__ __forceinline__ unsigned short f32_to_bf16(float f) {
    unsigned int u = __float_as_uint(f);
    u += 0x7FFFu + ((u >> 16) & 1u);   // round to nearest even
    return (unsigned short)(u >> 16);
}

__device__ __forceinline__ v8f wmma_bf16(const Frag& a, const Frag& b, v8f c) {
    return __builtin_amdgcn_wmma_f32_16x16x32_bf16(
        /*neg_a=*/false, a.v, /*neg_b=*/false, b.v,
        /*c_mod=*/(short)0, c, /*reuse_a=*/false, /*reuse_b=*/false);
}

// ---------------------------------------------------------------------------
// CDNA5 async global->LDS copy (GLOBAL_LOAD_ASYNC_TO_LDS_B128, ASYNCcnt-tracked)
// with a safe synchronous fallback if the builtin is unavailable.
// Builtin wants typed v4i pointers in AS(1)/AS(3); generic->AS conversion must
// go through an integer (C-style cast), as static_cast rejects it.
// ---------------------------------------------------------------------------
#if defined(__gfx1250__) && __has_builtin(__builtin_amdgcn_global_load_async_to_lds_b128)
#define ATHENA_HAS_ASYNC_LDS 1
#else
#define ATHENA_HAS_ASYNC_LDS 0
#endif

__device__ __forceinline__ void async_copy_b128(unsigned short* lds_dst,
                                                const unsigned short* gsrc) {
#if ATHENA_HAS_ASYNC_LDS
    __builtin_amdgcn_global_load_async_to_lds_b128(
        (v4i __attribute__((address_space(1)))*)(uintptr_t)gsrc,
        (v4i __attribute__((address_space(3)))*)(uintptr_t)lds_dst,
        /*imm offset=*/0, /*cpol=*/0);
#else
    *(uint4*)lds_dst = *(const uint4*)gsrc;
#endif
}

__device__ __forceinline__ void wait_async() {
#if defined(__gfx1250__) && __has_builtin(__builtin_amdgcn_s_wait_asynccnt)
    __builtin_amdgcn_s_wait_asynccnt(0);
#elif ATHENA_HAS_ASYNC_LDS
    asm volatile("s_wait_asynccnt 0x0" ::: "memory");
#endif
}

// ---------------------------------------------------------------------------
// Elementwise fp32 -> bf16
// ---------------------------------------------------------------------------
__global__ void cvt_f2b(const float* __restrict__ src,
                        unsigned short* __restrict__ dst, int n) {
    int i = blockIdx.x * blockDim.x + threadIdx.x;
    if (i < n) dst[i] = f32_to_bf16(src[i]);
}

// (H, D, DK) fp32 -> (D, H*DK) bf16   (H=8, D=1024, DK=128 hardcoded)
__global__ void cvt_qkvw(const float* __restrict__ src,
                         unsigned short* __restrict__ dst) {
    int i = blockIdx.x * blockDim.x + threadIdx.x;     // over 8*1024*128 = 1M
    if (i >= 8 * 1024 * 128) return;
    int kk = i & 127;
    int d  = (i >> 7) & 1023;
    int h  = i >> 17;
    dst[(size_t)d * 1024 + h * 128 + kk] = f32_to_bf16(src[i]);
}

// ---------------------------------------------------------------------------
// Generic tiled WMMA GEMM:  C[M,N] = A[M,K](bf16) x B[K,N](bf16) (+bias)(+resid)(+relu)
// BM=BN=128, BK=32. 256 threads = 8 waves, wave grid 2x4, wave tile 64x32.
// Double-buffered LDS pipeline: tile t+1 streams via async-to-LDS (A) and
// global->VGPR (B, transposed scatter) while tile t runs on the WMMA units.
// M,N multiples of 128; K multiple of 32.
// ---------------------------------------------------------------------------
template <bool OUTF32, bool RELU>
__global__ __launch_bounds__(256) void gemm_wmma(
    const unsigned short* __restrict__ A, const unsigned short* __restrict__ Bm,
    void* __restrict__ Cout, const float* __restrict__ bias,
    const float* __restrict__ resid, int Mdim, int Ndim, int Kdim) {
    __shared__ unsigned short As[2][128 * 40];   // row-major, stride 40 (pad 8)
    __shared__ unsigned short Bs[2][128 * 40];   // transposed [n][k], stride 40

    const int tid  = threadIdx.x;
    const int lane = tid & 31, w = tid >> 5;
    const int wm = w >> 2, wn = w & 3;
    const int hi = lane >> 4, l16 = lane & 15;
    const int bm = blockIdx.y * 128, bn = blockIdx.x * 128;

    // per-thread staging coordinates (512 x b128 per tile, 2 per thread)
    const int ar0 = tid >> 2,         ac0 = (tid & 3) * 8;           // A slot 0
    const int ar1 = (tid + 256) >> 2, ac1 = ((tid + 256) & 3) * 8;   // A slot 1
    const int bk0 = tid >> 4,         bc0 = (tid & 15) * 8;          // B slot 0
    const int bk1 = (tid + 256) >> 4, bc1 = ((tid + 256) & 15) * 8;  // B slot 1

    v8f acc[4][2];
#pragma unroll
    for (int mt = 0; mt < 4; ++mt)
#pragma unroll
        for (int nt = 0; nt < 2; ++nt) acc[mt][nt] = {};

    const int kbo = hi ? 8 : 0;     // A-fragment lane K offset
    const int kb2 = hi ? 16 : 0;    // unused (B fragments also use kbo)

    // ---- prologue: stage tile 0 into buffer 0 ----
    async_copy_b128(&As[0][ar0 * 40 + ac0], A + (size_t)(bm + ar0) * Kdim + ac0);
    async_copy_b128(&As[0][ar1 * 40 + ac1], A + (size_t)(bm + ar1) * Kdim + ac1);
    {
        BF8 b0, b1;
        b0.q = *(const uint4*)(Bm + (size_t)bk0 * Ndim + bn + bc0);
        b1.q = *(const uint4*)(Bm + (size_t)bk1 * Ndim + bn + bc1);
#pragma unroll
        for (int j = 0; j < 8; ++j) {
            Bs[0][(bc0 + j) * 40 + bk0] = b0.e[j];
            Bs[0][(bc1 + j) * 40 + bk1] = b1.e[j];
        }
    }
    wait_async();
    __syncthreads();

    int cur = 0;
    for (int kb = 0; kb < Kdim; kb += 32) {
        const int nxt = cur ^ 1;
        const bool has_next = (kb + 32 < Kdim);

        // ---- kick off tile t+1: async A -> LDS, B -> VGPRs ----
        BF8 b0, b1;
        if (has_next) {
            const int kn = kb + 32;
            async_copy_b128(&As[nxt][ar0 * 40 + ac0],
                            A + (size_t)(bm + ar0) * Kdim + kn + ac0);
            async_copy_b128(&As[nxt][ar1 * 40 + ac1],
                            A + (size_t)(bm + ar1) * Kdim + kn + ac1);
            b0.q = *(const uint4*)(Bm + (size_t)(kn + bk0) * Ndim + bn + bc0);
            b1.q = *(const uint4*)(Bm + (size_t)(kn + bk1) * Ndim + bn + bc1);
        }

        // ---- compute on tile t ----
        Frag af[4], bfr[2];
#pragma unroll
        for (int mt = 0; mt < 4; ++mt) {
            int r = wm * 64 + mt * 16 + l16;
            af[mt].q[0] = *(const uint4*)&As[cur][r * 40 + kbo];
            af[mt].q[1] = *(const uint4*)&As[cur][r * 40 + 16 + kbo];
        }
#pragma unroll
        for (int nt = 0; nt < 2; ++nt) {
            int n = wn * 32 + nt * 16 + l16;
            bfr[nt].q[0] = *(const uint4*)&Bs[cur][n * 40 + kbo];
            bfr[nt].q[1] = *(const uint4*)&Bs[cur][n * 40 + 16 + kbo];
        }
#pragma unroll
        for (int mt = 0; mt < 4; ++mt)
#pragma unroll
            for (int nt = 0; nt < 2; ++nt)
                acc[mt][nt] = wmma_bf16(af[mt], bfr[nt], acc[mt][nt]);

        // ---- finish staging tile t+1, single barrier per K-step ----
        if (has_next) {
#pragma unroll
            for (int j = 0; j < 8; ++j) {
                Bs[nxt][(bc0 + j) * 40 + bk0] = b0.e[j];
                Bs[nxt][(bc1 + j) * 40 + bk1] = b1.e[j];
            }
            wait_async();
        }
        __syncthreads();
        cur = nxt;
    }

    // ---- epilogue: bias / residual / relu / store ----
#pragma unroll
    for (int mt = 0; mt < 4; ++mt)
#pragma unroll
        for (int nt = 0; nt < 2; ++nt) {
            int col  = bn + wn * 32 + nt * 16 + l16;
            float bi = bias ? bias[col] : 0.0f;
#pragma unroll
            for (int v = 0; v < 8; ++v) {
                int row   = bm + wm * 64 + mt * 16 + v + 8 * hi;
                float val = acc[mt][nt][v] + bi;
                if (resid) val += resid[(size_t)row * Ndim + col];
                if (RELU) val = fmaxf(val, 0.0f);
                if (OUTF32)
                    ((float*)Cout)[(size_t)row * Ndim + col] = val;
                else
                    ((unsigned short*)Cout)[(size_t)row * Ndim + col] = f32_to_bf16(val);
            }
        }
}

// ---------------------------------------------------------------------------
// Flash attention: per block = (b, h, 64-query tile), 128 threads = 4 waves,
// each wave owns 16 query rows. Key tiles of 64. DK=DV=128, S=2048, H=8.
// q/k/v/o layout: [B, S, H, 128] bf16.  Mask per reference: m*A + (1-m)*(-1e-30).
// Q and K tiles stream via async-to-LDS; V uses a register transpose.
// ---------------------------------------------------------------------------
__global__ __launch_bounds__(128) void attn_kernel(
    const unsigned short* __restrict__ qb, const unsigned short* __restrict__ kbuf,
    const unsigned short* __restrict__ vbuf, const int* __restrict__ mask,
    unsigned short* __restrict__ ob) {
    const int S = 2048, H = 8, DH = 128;
    const int qt = blockIdx.x, h = blockIdx.y, b = blockIdx.z;
    const int tid = threadIdx.x, lane = tid & 31, w = tid >> 5;
    const int hi = lane >> 4, l16 = lane & 15;

    __shared__ unsigned short Qs[64 * 136];   // [qrow][dk]
    __shared__ unsigned short KV[128 * 72];   // K phase: [key][dk] s=136; V phase: [dv][key] s=72
    __shared__ unsigned short Ps[4 * 16 * 72];// per-wave P staging [16][keys64] s=72

    const int q0 = qt * 64;
    const unsigned short* qg = qb + ((size_t)b * S) * H * DH + (size_t)h * DH;
    const unsigned short* kg = kbuf + ((size_t)b * S) * H * DH + (size_t)h * DH;
    const unsigned short* vg = vbuf + ((size_t)b * S) * H * DH + (size_t)h * DH;
    const int* mp = mask + b * S;

    // load Q tile (64 x 128) via async-to-LDS
#pragma unroll
    for (int i = 0; i < 8; ++i) {
        int u = tid + i * 128;
        int r = u >> 4, c = (u & 15) * 8;
        async_copy_b128(&Qs[r * 136 + c], qg + (size_t)(q0 + r) * H * DH + c);
    }

    v8f ot[8];
#pragma unroll
    for (int dt = 0; dt < 8; ++dt) ot[dt] = {};
    float mrow[8], lrow[8];
#pragma unroll
    for (int v = 0; v < 8; ++v) { mrow[v] = -3.0e38f; lrow[v] = 0.0f; }

    const int kbo = hi ? 8 : 0, kb2 = hi ? 16 : 0;
    wait_async();
    __syncthreads();

    for (int kt = 0; kt < S / 64; ++kt) {
        // ---- load K tile [64 keys][128 dk] via async-to-LDS ----
#pragma unroll
        for (int i = 0; i < 8; ++i) {
            int u = tid + i * 128;
            int r = u >> 4, c = (u & 15) * 8;
            async_copy_b128(&KV[r * 136 + c], kg + (size_t)(kt * 64 + r) * H * DH + c);
        }
        wait_async();
        __syncthreads();

        // ---- scores: S = Q x K^T (16 rows x 64 keys per wave) ----
        v8f sc[4];
#pragma unroll
        for (int nt = 0; nt < 4; ++nt) sc[nt] = {};
#pragma unroll
        for (int ks = 0; ks < 4; ++ks) {
            const int kk = ks * 32;
            Frag af;
            int r = w * 16 + l16;
            af.q[0] = *(const uint4*)&Qs[r * 136 + kk + kbo];
            af.q[1] = *(const uint4*)&Qs[r * 136 + kk + 16 + kbo];
#pragma unroll
            for (int nt = 0; nt < 4; ++nt) {
                Frag bf_;
                int n = nt * 16 + l16;
                bf_.q[0] = *(const uint4*)&KV[n * 136 + kk + kbo];
                bf_.q[1] = *(const uint4*)&KV[n * 136 + kk + 16 + kbo];
                sc[nt] = wmma_bf16(af, bf_, sc[nt]);
            }
        }
        __syncthreads();   // all waves done reading K

        // ---- masking (reference semantics) ----
#pragma unroll
        for (int nt = 0; nt < 4; ++nt) {
            float mv = (float)mp[kt * 64 + nt * 16 + l16];
#pragma unroll
            for (int v = 0; v < 8; ++v)
                sc[nt][v] = mv * sc[nt][v] + (1.0f - mv) * (-1e-30f);
        }

        // ---- online softmax per query row (16-lane half-wave reductions) ----
        float mnew[8];
#pragma unroll
        for (int v = 0; v < 8; ++v) {
            float t = fmaxf(fmaxf(sc[0][v], sc[1][v]), fmaxf(sc[2][v], sc[3][v]));
            t = fmaxf(t, __shfl_xor(t, 1));
            t = fmaxf(t, __shfl_xor(t, 2));
            t = fmaxf(t, __shfl_xor(t, 4));
            t = fmaxf(t, __shfl_xor(t, 8));
            mnew[v] = fmaxf(mrow[v], t);
        }
#pragma unroll
        for (int v = 0; v < 8; ++v) {
            float sum = 0.0f;
#pragma unroll
            for (int nt = 0; nt < 4; ++nt) {
                float p = __expf(sc[nt][v] - mnew[v]);
                sc[nt][v] = p;
                sum += p;
            }
            sum += __shfl_xor(sum, 1);
            sum += __shfl_xor(sum, 2);
            sum += __shfl_xor(sum, 4);
            sum += __shfl_xor(sum, 8);
            float scale = __expf(mrow[v] - mnew[v]);
            lrow[v] = lrow[v] * scale + sum;
            mrow[v] = mnew[v];
#pragma unroll
            for (int dt = 0; dt < 8; ++dt) ot[dt][v] *= scale;
        }

        // ---- P -> per-wave LDS staging (C-layout -> A-layout reshape) ----
#pragma unroll
        for (int nt = 0; nt < 4; ++nt)
#pragma unroll
            for (int v = 0; v < 8; ++v)
                Ps[(w * 16 + v + 8 * hi) * 72 + nt * 16 + l16] = f32_to_bf16(sc[nt][v]);

        // ---- load V tile transposed: KV becomes [dv][key], stride 72 ----
#pragma unroll
        for (int i = 0; i < 8; ++i) {
            int u = tid + i * 128;
            int key = u >> 4, d0 = (u & 15) * 8;
            BF8 d;
            d.q = *(const uint4*)(vg + (size_t)(kt * 64 + key) * H * DH + d0);
#pragma unroll
            for (int j = 0; j < 8; ++j) KV[(d0 + j) * 72 + key] = d.e[j];
        }
        __syncthreads();

        // ---- O += P(16x64) x V(64x128) ----
#pragma unroll
        for (int ks = 0; ks < 2; ++ks) {
            Frag af;
            af.q[0] = *(const uint4*)&Ps[(w * 16 + l16) * 72 + ks * 32 + kbo];
            af.q[1] = *(const uint4*)&Ps[(w * 16 + l16) * 72 + ks * 32 + 16 + kbo];
#pragma unroll
            for (int dt = 0; dt < 8; ++dt) {
                Frag bf_;
                bf_.q[0] = *(const uint4*)&KV[(dt * 16 + l16) * 72 + ks * 32 + kbo];
                bf_.q[1] = *(const uint4*)&KV[(dt * 16 + l16) * 72 + ks * 32 + 16 + kbo];
                ot[dt] = wmma_bf16(af, bf_, ot[dt]);
            }
        }
        __syncthreads();   // before KV is overwritten next iteration
    }

    // ---- normalize + store O (bf16, [B,S,H,DV]) ----
#pragma unroll
    for (int dt = 0; dt < 8; ++dt)
#pragma unroll
        for (int v = 0; v < 8; ++v) {
            int row = q0 + w * 16 + v + 8 * hi;
            float val = ot[dt][v] / lrow[v];
            ob[(((size_t)b * S + row) * H + h) * DH + dt * 16 + l16] = f32_to_bf16(val);
        }
}

// ---------------------------------------------------------------------------
// LayerNorm over D=1024: one block per row; writes fp32 + bf16 mirrors.
// ---------------------------------------------------------------------------
__global__ __launch_bounds__(256) void layernorm_kernel(
    const float* __restrict__ src, float* __restrict__ dstf,
    unsigned short* __restrict__ dstb, const float* __restrict__ gamma,
    const float* __restrict__ beta, int gidx) {
    const int row = blockIdx.x, tid = threadIdx.x;
    const float* x = src + (size_t)row * 1024;
    float4 xv = *(const float4*)(x + tid * 4);
    float s  = xv.x + xv.y + xv.z + xv.w;
    float s2 = xv.x * xv.x + xv.y * xv.y + xv.z * xv.z + xv.w * xv.w;
#pragma unroll
    for (int off = 1; off < 32; off <<= 1) {
        s  += __shfl_xor(s, off);
        s2 += __shfl_xor(s2, off);
    }
    __shared__ float rs[8], rs2[8];
    int w = tid >> 5, lane = tid & 31;
    if (lane == 0) { rs[w] = s; rs2[w] = s2; }
    __syncthreads();
    if (tid == 0) {
        float a = 0.f, b2 = 0.f;
#pragma unroll
        for (int i = 0; i < 8; ++i) { a += rs[i]; b2 += rs2[i]; }
        rs[0] = a; rs2[0] = b2;
    }
    __syncthreads();
    float mean = rs[0] * (1.0f / 1024.0f);
    float var  = rs2[0] * (1.0f / 1024.0f) - mean * mean;
    float inv  = rsqrtf(var + 1e-14f);
    float g = gamma[gidx], bt = beta[gidx];
    float4 y;
    y.x = (xv.x - mean) * inv * g + bt;
    y.y = (xv.y - mean) * inv * g + bt;
    y.z = (xv.z - mean) * inv * g + bt;
    y.w = (xv.w - mean) * inv * g + bt;
    *(float4*)(dstf + (size_t)row * 1024 + tid * 4) = y;
    uint2 p;
    p.x = (unsigned)f32_to_bf16(y.x) | ((unsigned)f32_to_bf16(y.y) << 16);
    p.y = (unsigned)f32_to_bf16(y.z) | ((unsigned)f32_to_bf16(y.w) << 16);
    *(uint2*)(dstb + (size_t)row * 1024 + tid * 4) = p;
}

// ---------------------------------------------------------------------------
// Host orchestration
// ---------------------------------------------------------------------------
extern "C" void kernel_launch(void* const* d_in, const int* in_sizes, int n_in,
                              void* d_out, int out_size, void* d_ws, size_t ws_size,
                              hipStream_t stream) {
    (void)in_sizes; (void)n_in; (void)out_size; (void)ws_size;
    const float* x    = (const float*)d_in[0];
    const int*   msk  = (const int*)d_in[1];
    const float* Wq   = (const float*)d_in[2];
    const float* Wk   = (const float*)d_in[3];
    const float* Wv   = (const float*)d_in[4];
    const float* Wo   = (const float*)d_in[5];
    const float* W1   = (const float*)d_in[6];
    const float* b1   = (const float*)d_in[7];
    const float* W2   = (const float*)d_in[8];
    const float* b2   = (const float*)d_in[9];
    const float* gamma = (const float*)d_in[10];
    const float* beta  = (const float*)d_in[11];

    const int Bc = 4, Sc = 2048, Dc = 1024, Hc = 8, DKc = 128, DFFc = 2048;
    const size_t Mrows = (size_t)Bc * Sc;   // 8192
    float* xf = (float*)d_out;              // fp32 running activation lives in d_out

    char* ws = (char*)d_ws;
    size_t off = 0;
    auto alloc = [&](size_t bytes) {
        size_t o = off;
        off += (bytes + 255) & ~(size_t)255;
        return ws + o;
    };
    unsigned short* xb  = (unsigned short*)alloc(Mrows * Dc * 2);
    unsigned short* qb  = (unsigned short*)alloc(Mrows * Hc * DKc * 2);
    unsigned short* kbf = (unsigned short*)alloc(Mrows * Hc * DKc * 2);
    unsigned short* vbf = (unsigned short*)alloc(Mrows * Hc * DKc * 2);
    unsigned short* obf = (unsigned short*)alloc(Mrows * Hc * DKc * 2);
    unsigned short* h1  = (unsigned short*)alloc(Mrows * DFFc * 2);
    float*          tmp = (float*)alloc(Mrows * Dc * 4);
    unsigned short* wqb = (unsigned short*)alloc((size_t)Dc * Hc * DKc * 2);
    unsigned short* wkb = (unsigned short*)alloc((size_t)Dc * Hc * DKc * 2);
    unsigned short* wvb = (unsigned short*)alloc((size_t)Dc * Hc * DKc * 2);
    unsigned short* wob = (unsigned short*)alloc((size_t)Hc * DKc * Dc * 2);
    unsigned short* w1b = (unsigned short*)alloc((size_t)Dc * DFFc * 2);
    unsigned short* w2b = (unsigned short*)alloc((size_t)DFFc * Dc * 2);

    // xf <- x (fp32), xb <- bf16(x)
    (void)hipMemcpyAsync(xf, x, Mrows * Dc * sizeof(float), hipMemcpyDeviceToDevice, stream);
    {
        int n = (int)(Mrows * Dc);
        cvt_f2b<<<(n + 255) / 256, 256, 0, stream>>>(x, xb, n);
    }

    dim3 blk(256);
    dim3 g1(1024 / 128, (unsigned)(Mrows / 128));   // N=1024 GEMMs
    dim3 g2(2048 / 128, (unsigned)(Mrows / 128));   // N=2048 GEMM
    dim3 ga(Sc / 64, Hc, Bc);

    for (int i = 0; i < 2; ++i) {
        const int nqkv = Hc * Dc * DKc;             // 1M
        cvt_qkvw<<<(nqkv + 255) / 256, 256, 0, stream>>>(Wq + (size_t)i * nqkv, wqb);
        cvt_qkvw<<<(nqkv + 255) / 256, 256, 0, stream>>>(Wk + (size_t)i * nqkv, wkb);
        cvt_qkvw<<<(nqkv + 255) / 256, 256, 0, stream>>>(Wv + (size_t)i * nqkv, wvb);
        const int nwo = Hc * DKc * Dc;              // 1M
        cvt_f2b<<<(nwo + 255) / 256, 256, 0, stream>>>(Wo + (size_t)i * nwo, wob, nwo);
        const int nw12 = Dc * DFFc;                 // 2M
        cvt_f2b<<<(nw12 + 255) / 256, 256, 0, stream>>>(W1 + (size_t)i * nw12, w1b, nw12);
        cvt_f2b<<<(nw12 + 255) / 256, 256, 0, stream>>>(W2 + (size_t)i * nw12, w2b, nw12);

        // Q/K/V projections -> [B,S,H,DK] bf16
        gemm_wmma<false, false><<<g1, blk, 0, stream>>>(xb, wqb, qb, nullptr, nullptr,
                                                        (int)Mrows, 1024, 1024);
        gemm_wmma<false, false><<<g1, blk, 0, stream>>>(xb, wkb, kbf, nullptr, nullptr,
                                                        (int)Mrows, 1024, 1024);
        gemm_wmma<false, false><<<g1, blk, 0, stream>>>(xb, wvb, vbf, nullptr, nullptr,
                                                        (int)Mrows, 1024, 1024);
        // attention
        attn_kernel<<<ga, 128, 0, stream>>>(qb, kbf, vbf, msk, obf);
        // output projection + residual (fp32)
        gemm_wmma<true, false><<<g1, blk, 0, stream>>>(obf, wob, tmp, nullptr, xf,
                                                       (int)Mrows, 1024, 1024);
        layernorm_kernel<<<(unsigned)Mrows, 256, 0, stream>>>(tmp, xf, xb, gamma, beta, 2 * i);
        // FFN
        gemm_wmma<false, true><<<g2, blk, 0, stream>>>(xb, w1b, h1, b1 + (size_t)i * DFFc,
                                                       nullptr, (int)Mrows, 2048, 1024);
        gemm_wmma<true, false><<<g1, blk, 0, stream>>>(h1, w2b, tmp, b2 + (size_t)i * Dc,
                                                       xf, (int)Mrows, 1024, 2048);
        layernorm_kernel<<<(unsigned)Mrows, 256, 0, stream>>>(tmp, xf, xb, gamma, beta, 2 * i + 1);
    }
}
